// TpMLPBlock_86655260164433
// MI455X (gfx1250) — matrix operand order repaired and checked by reading, not compiled, not run on hardware
//
#include <hip/hip_runtime.h>
#include <hip/hip_bf16.h>

// ---------------------------------------------------------------------------
// TpMLP block for MI455X (gfx1250, wave32, WMMA).
// All GEMMs run on v_wmma_f32_16x16x32_bf16 (f32 accumulate, bf16 operands).
// Workspace requirement: ~275 MB.
// ---------------------------------------------------------------------------

typedef __attribute__((ext_vector_type(16))) __bf16 v16bf;
typedef __attribute__((ext_vector_type(8)))  float  v8f;

union BfVec {
  v16bf v;
  __bf16 h[16];
  unsigned short s[16];
  uint4 q[2];
};

// native RNE conversions (lets the backend pick hardware bf16 cvt ops)
__device__ __forceinline__ unsigned short f2bf(float f) {
  __bf16 b = (__bf16)f;
  return __builtin_bit_cast(unsigned short, b);
}
__device__ __forceinline__ float bf2f(unsigned short u) {
  return (float)__builtin_bit_cast(__bf16, u);
}

__device__ __forceinline__ v8f wmma_bf16(v16bf a, v16bf b, v8f c) {
  return __builtin_amdgcn_wmma_f32_16x16x32_bf16(
      false, a, false, b, (short)0, c, false, false);
}

// u-order pixel p (tp1 space): bb=p>>10, hh=(p>>5)&31, ww=p&31,
// bb = b*4 + n1*2 + n2  ->  x pixel (b, 2hh+n1, 2ww+n2)
__device__ __forceinline__ int u2x_pixel(int p) {
  int bb = p >> 10, hh = (p >> 5) & 31, ww = p & 31;
  int b = bb >> 2, n1 = (bb >> 1) & 1, n2 = bb & 1;
  return (b * 64 + (hh * 2 + n1)) * 64 + (ww * 2 + n2);
}

// pack 8 consecutive f32 -> BfVec elements [base..base+7]
__device__ __forceinline__ void pack8(BfVec& a, int base, const float* ptr) {
  float4 f0 = ((const float4*)ptr)[0];
  float4 f1 = ((const float4*)ptr)[1];
  a.h[base+0]=(__bf16)f0.x; a.h[base+1]=(__bf16)f0.y;
  a.h[base+2]=(__bf16)f0.z; a.h[base+3]=(__bf16)f0.w;
  a.h[base+4]=(__bf16)f1.x; a.h[base+5]=(__bf16)f1.y;
  a.h[base+6]=(__bf16)f1.z; a.h[base+7]=(__bf16)f1.w;
}

// ------------------------- K0: weight prep ---------------------------------
__global__ void k_prep(const float* mh_pin, const float* mw_pin,
                       const float* mh_full, const float* mw_full,
                       const float* mh_pout, const float* mw_pout,
                       const float* proj_w,
                       unsigned short* pinT,   // [32][512]  (n<16: mh, else mw)
                       unsigned short* fullH,  // [128][1024]
                       unsigned short* fullW,  // [128][1024]
                       unsigned short* poutT,  // [2][512][16] transposed
                       unsigned short* pwT,    // [512][512] transposed
                       float* a_sum)           // zeroed [16*512]
{
  int i = blockIdx.x * blockDim.x + threadIdx.x;
  int stride = gridDim.x * blockDim.x;
  for (int t = i; t < 32 * 512; t += stride) {
    int n = t >> 9, ch = t & 511;
    float v = (n < 16) ? mh_pin[ch * 16 + n] : mw_pin[ch * 16 + (n - 16)];
    pinT[t] = f2bf(v);
  }
  for (int t = i; t < 128 * 1024; t += stride) {
    fullH[t] = f2bf(mh_full[t]);
    fullW[t] = f2bf(mw_full[t]);
  }
  for (int t = i; t < 2 * 512 * 16; t += stride) {
    int path = t >> 13, r = t & 8191;
    int n = r >> 4, k = r & 15;
    const float* src = path ? mw_pout : mh_pout;
    poutT[t] = f2bf(src[k * 512 + n]);
  }
  for (int t = i; t < 512 * 512; t += stride) {
    int n = t >> 9, k = t & 511;
    pwT[t] = f2bf(proj_w[k * 512 + n]);
  }
  for (int t = i; t < 16 * 512; t += stride) a_sum[t] = 0.0f;
}

// ------------------------- K1: fused proj_in -------------------------------
// t_{h,w}[p][16] = u[p][512] @ pin_w + pin_b   (u-order rows)
__global__ void __launch_bounds__(64)
k_proj_in(const float* __restrict__ x,
          const unsigned short* __restrict__ pinT,
          const float* mh_pin_b, const float* mw_pin_b,
          float* __restrict__ t_h, float* __restrict__ t_w)
{
  int rt   = blockIdx.x;            // 0..4095 row tiles
  int wave = threadIdx.x >> 5;      // 0 = h path, 1 = w path
  int L    = threadIdx.x & 31;
  int p0   = rt * 16;
  int lane_p = p0 + (L & 15);
  const float* xr = x + (long)u2x_pixel(lane_p) * 512;
  int selA = (L < 16) ? 0 : 8;
  int bsel = (L < 16) ? 0 : 16;
  const unsigned short* brow = pinT + ((wave * 16 + (L & 15)) * 512);
  v8f acc = {};
  for (int kt = 0; kt < 16; ++kt) {
    int kb = kt * 32;
    BfVec a;
    pack8(a, 0, xr + kb + selA);
    pack8(a, 8, xr + kb + 16 + selA);
    BfVec bm;
    const uint4* bp = (const uint4*)(brow + kb + selA);
    bm.q[0] = bp[0]; bm.q[1] = bp[2];
    acc = wmma_bf16(a.v, bm.v, acc);
  }
  const float* pbias = wave ? mw_pin_b : mh_pin_b;
  float* dst = wave ? t_w : t_h;
  int col = L & 15;
  float bias = pbias[col];
  int rbase = (L < 16) ? 0 : 8;
  for (int j = 0; j < 8; ++j)
    dst[(p0 + rbase + j) * 16 + col] = acc[j] + bias;
}

// ------------------------- K2: windowed mixing -----------------------------
// per head m: z[win][f] = sum_d1 t[win][d1] * full_w[f][m*128+d1] + full_b[f]
// mode 0: mix_h (H_sp=32,W_sp=2)   mode 1: mix_w (H_sp=2,W_sp=32)
__device__ __forceinline__ void mix_decode(int mode, int m, int win, int f,
                                           int& p, int& ch) {
  int bb = win >> 4;
  if (mode == 0) {
    int wc = win & 15;
    int hh = f >> 2, wl = (f >> 1) & 1, d = f & 1;
    p = (bb << 10) + (hh << 5) + (wc * 2 + wl);
    ch = m * 2 + d;
  } else {
    int hr = win & 15;
    int wwp = (f & 63) >> 1, d = f & 1, hb = f >> 6;
    p = (bb << 10) + ((hr * 2 + hb) << 5) + wwp;
    ch = m * 2 + d;
  }
}

__global__ void __launch_bounds__(128)
k_mix(const float* __restrict__ t_src,
      const unsigned short* __restrict__ full_bf,
      const float* __restrict__ full_b,
      float* __restrict__ z_dst, int mode)
{
  int rt   = blockIdx.x;                 // 0..63 (window tiles)
  int wave = threadIdx.x >> 5;           // 4 waves/block
  int L    = threadIdx.x & 31;
  int ct   = blockIdx.y * 4 + wave;      // 0..63
  int m     = ct >> 3;
  int fbase = (ct & 7) * 16;
  int w0    = rt * 16;
  int win   = w0 + (L & 15);
  int selA  = (L < 16) ? 0 : 8;
  int bsel  = (L < 16) ? 0 : 16;
  v8f acc = {};
  for (int kt = 0; kt < 4; ++kt) {
    int kb = kt * 32;
    BfVec a;
    #pragma unroll
    for (int e = 0; e < 16; ++e) {
      int d1 = kb + ((e < 8) ? (selA + e) : (16 + selA + e - 8));
      int p, ch;
      mix_decode(mode, m, win, d1, p, ch);
      a.h[e] = (__bf16)t_src[p * 16 + ch];
    }
    BfVec bm;
    int f = fbase + (L & 15);
    const uint4* bp = (const uint4*)(full_bf + (f * 1024 + m * 128 + kb + selA));
    bm.q[0] = bp[0]; bm.q[1] = bp[2];
    acc = wmma_bf16(a.v, bm.v, acc);
  }
  int colf = fbase + (L & 15);
  float bias = full_b[colf];
  int rbase = (L < 16) ? 0 : 8;
  for (int j = 0; j < 8; ++j) {
    int wino = w0 + rbase + j;
    int p, ch;
    mix_decode(mode, m, wino, colf, p, ch);
    z_dst[p * 16 + ch] = acc[j] + bias;
  }
}

// ------------------------- K3: fused proj_out ------------------------------
// {h,w}_bf[p][512] = z[p][16] @ pout_w + pout_b  (bf16 store, u-order)
__global__ void __launch_bounds__(64)
k_proj_out(const float* __restrict__ z_h,
           const float* __restrict__ z_w,
           const unsigned short* __restrict__ poutT,
           const float* mh_pout_b, const float* mw_pout_b,
           unsigned short* __restrict__ h_bf,
           unsigned short* __restrict__ w_bf)
{
  int rt   = blockIdx.x;             // 4096 row tiles
  int wave = threadIdx.x >> 5;       // 0 = h, 1 = w
  int L    = threadIdx.x & 31;
  const float* z = wave ? z_w : z_h;
  const unsigned short* pT = poutT + wave * (512 * 16);
  const float* pb = wave ? mw_pout_b : mh_pout_b;
  unsigned short* dst = wave ? w_bf : h_bf;
  int p0 = rt * 16;
  int pr = p0 + (L & 15);
  BfVec a;
  {
    int sel = (L < 16) ? 0 : 8;       // K pad: real K = 0..15, rest zero
    pack8(a, 0, z + pr * 16 + sel);
    #pragma unroll
    for (int e = 8; e < 16; ++e) a.s[e] = 0;
  }
  int rbase = (L < 16) ? 0 : 8;
  for (int ct = 0; ct < 32; ++ct) {
    BfVec bm;
    {
      const uint4* bp = (const uint4*)(pT + (ct * 16 + (L & 15)) * 16 + ((L < 16) ? 0 : 8));
      bm.q[0] = bp[0];
      bm.q[1] = make_uint4(0, 0, 0, 0);
    }
    v8f acc = {};
    acc = wmma_bf16(a.v, bm.v, acc);
    int col = ct * 16 + (L & 15);
    float bias = pb[col];
    for (int j = 0; j < 8; ++j)
      dst[(long)(p0 + rbase + j) * 512 + col] = f2bf(acc[j] + bias);
  }
}

// ------------------------- K4: MixC ----------------------------------------
__global__ void k_mixc(const float* __restrict__ x,
                       const float* __restrict__ c_weight,
                       const float* __restrict__ c_bias,
                       unsigned short* __restrict__ c_bf)
{
  int idx = blockIdx.x * blockDim.x + threadIdx.x;
  if (idx >= 16 * 32 * 32 * 512) return;
  int ch = idx & 511;
  int blk = idx >> 9;
  int b = blk >> 10, h2 = (blk >> 5) & 31, w2 = blk & 31;
  float xv[4];
  #pragma unroll
  for (int q = 0; q < 4; ++q) {
    int X = (b * 64 + (2 * h2 + (q >> 1))) * 64 + (2 * w2 + (q & 1));
    xv[q] = x[(long)X * 512 + ch];
  }
  #pragma unroll
  for (int qo = 0; qo < 4; ++qo) {
    float acc = c_bias[ch * 4 + qo];
    #pragma unroll
    for (int q = 0; q < 4; ++q) acc += xv[q] * c_weight[ch * 16 + q * 4 + qo];
    int bb = b * 4 + qo;
    int p = (bb << 10) + (h2 << 5) + w2;
    c_bf[(long)p * 512 + ch] = f2bf(acc);
  }
}

// ------------------------- K5: spatial mean accumulate ---------------------
__global__ void k_mean(const unsigned short* __restrict__ h_bf,
                       const unsigned short* __restrict__ w_bf,
                       const unsigned short* __restrict__ c_bf,
                       float* __restrict__ a_sum)
{
  int b = blockIdx.x >> 4, slice = blockIdx.x & 15;
  int ch = threadIdx.x;                  // 512 threads
  float s = 0.f;
  for (int q = 0; q < 4; ++q) {
    int bb = b * 4 + q;
    for (int i = 0; i < 64; ++i) {
      int p = (bb << 10) + slice * 64 + i;
      long o = (long)p * 512 + ch;
      s += bf2f(h_bf[o]) + bf2f(w_bf[o]) + bf2f(c_bf[o]);
    }
  }
  atomicAdd(&a_sum[b * 512 + ch], s);
}

// ------------------------- K6: reweight MLP + softmax ----------------------
__global__ void k_reweight(const float* __restrict__ a_sum,
                           const float* __restrict__ fc1_w, const float* fc1_b,
                           const float* __restrict__ fc2_w, const float* fc2_b,
                           float* __restrict__ wgt)  // [3][16][512]
{
  __shared__ float hid[128];
  __shared__ float logit[1536];
  int b = blockIdx.x;
  int t = threadIdx.x;                 // 128 threads
  float acc = fc1_b[t];
  for (int ch = 0; ch < 512; ++ch)
    acc += (a_sum[b * 512 + ch] * (1.0f / 4096.0f)) * fc1_w[ch * 128 + t];
  hid[t] = 0.5f * acc * (1.0f + erff(acc * 0.7071067811865476f));
  __syncthreads();
  for (int g = t; g < 1536; g += 128) {
    float a2 = fc2_b[g];
    for (int k = 0; k < 128; ++k) a2 += hid[k] * fc2_w[k * 1536 + g];
    logit[g] = a2;
  }
  __syncthreads();
  for (int c = t; c < 512; c += 128) {
    float l0 = logit[c * 3 + 0], l1 = logit[c * 3 + 1], l2 = logit[c * 3 + 2];
    float mx = fmaxf(l0, fmaxf(l1, l2));
    float e0 = __expf(l0 - mx), e1 = __expf(l1 - mx), e2 = __expf(l2 - mx);
    float inv = 1.0f / (e0 + e1 + e2);
    wgt[(0 * 16 + b) * 512 + c] = e0 * inv;
    wgt[(1 * 16 + b) * 512 + c] = e1 * inv;
    wgt[(2 * 16 + b) * 512 + c] = e2 * inv;
  }
}

// ------------------------- K7: combine y = h*a0 + w*a1 + c*a2 --------------
__global__ void k_combine(const unsigned short* __restrict__ h_bf,
                          const unsigned short* __restrict__ w_bf,
                          const unsigned short* __restrict__ c_bf,
                          const float* __restrict__ wgt,
                          unsigned short* __restrict__ y_bf)
{
  long idx = (long)blockIdx.x * blockDim.x + threadIdx.x;
  if (idx >= (long)65536 * 512) return;
  int ch = (int)(idx & 511);
  int p = (int)(idx >> 9);
  int b = p >> 12;            // bb = p>>10, b = bb>>2
  float v = bf2f(h_bf[idx]) * wgt[(0 * 16 + b) * 512 + ch]
          + bf2f(w_bf[idx]) * wgt[(1 * 16 + b) * 512 + ch]
          + bf2f(c_bf[idx]) * wgt[(2 * 16 + b) * 512 + ch];
  y_bf[idx] = f2bf(v);
}

// ------------------------- K8: final projection GEMM -----------------------
// out[xpix][512] = y[p][512] @ proj_w + proj_b, tp2 applied on store
__global__ void __launch_bounds__(128)
k_final(const unsigned short* __restrict__ y_bf,
        const unsigned short* __restrict__ pwT,
        const float* __restrict__ proj_b,
        float* __restrict__ out)
{
  int rt   = blockIdx.x;             // 4096 row tiles
  int wave = threadIdx.x >> 5;       // 4 waves/block
  int L    = threadIdx.x & 31;
  int cg   = blockIdx.y * 4 + wave;  // 0..7 col groups (4 tiles each)
  int p0   = rt * 16;
  int pr   = p0 + (L & 15);
  int selA = (L < 16) ? 0 : 8;
  int bsel = (L < 16) ? 0 : 16;
  const unsigned short* yrow = y_bf + (long)pr * 512;
  v8f accs[4] = {};
  for (int kt = 0; kt < 16; ++kt) {
    int kb = kt * 32;
    BfVec a;
    a.q[0] = *(const uint4*)(yrow + kb + selA);
    a.q[1] = *(const uint4*)(yrow + kb + 16 + selA);
    #pragma unroll
    for (int cc = 0; cc < 4; ++cc) {
      int N = (cg * 4 + cc) * 16 + (L & 15);
      BfVec bm;
      const uint4* bp = (const uint4*)(pwT + ((long)N * 512 + kb + selA));
      bm.q[0] = bp[0]; bm.q[1] = bp[2];
      accs[cc] = wmma_bf16(a.v, bm.v, accs[cc]);
    }
  }
  int rbase = (L < 16) ? 0 : 8;
  #pragma unroll
  for (int cc = 0; cc < 4; ++cc) {
    int col = (cg * 4 + cc) * 16 + (L & 15);
    float bias = proj_b[col];
    for (int j = 0; j < 8; ++j) {
      int p = p0 + rbase + j;
      long xp = (long)u2x_pixel(p);
      out[xp * 512 + col] = accs[cc][j] + bias;
    }
  }
}

// ---------------------------------------------------------------------------
extern "C" void kernel_launch(void* const* d_in, const int* in_sizes, int n_in,
                              void* d_out, int out_size, void* d_ws, size_t ws_size,
                              hipStream_t stream) {
  const float* x          = (const float*)d_in[0];
  const float* mh_pin_w   = (const float*)d_in[1];
  const float* mh_pin_b   = (const float*)d_in[2];
  const float* mh_full_w  = (const float*)d_in[3];
  const float* mh_full_b  = (const float*)d_in[4];
  const float* mh_pout_w  = (const float*)d_in[5];
  const float* mh_pout_b  = (const float*)d_in[6];
  const float* mw_pin_w   = (const float*)d_in[7];
  const float* mw_pin_b   = (const float*)d_in[8];
  const float* mw_full_w  = (const float*)d_in[9];
  const float* mw_full_b  = (const float*)d_in[10];
  const float* mw_pout_w  = (const float*)d_in[11];
  const float* mw_pout_b  = (const float*)d_in[12];
  const float* c_weight   = (const float*)d_in[13];
  const float* c_bias     = (const float*)d_in[14];
  const float* rw_fc1_w   = (const float*)d_in[15];
  const float* rw_fc1_b   = (const float*)d_in[16];
  const float* rw_fc2_w   = (const float*)d_in[17];
  const float* rw_fc2_b   = (const float*)d_in[18];
  const float* proj_w     = (const float*)d_in[19];
  const float* proj_b     = (const float*)d_in[20];
  float* out = (float*)d_out;

  char* ws = (char*)d_ws;
  size_t off = 0;
  auto take = [&](size_t bytes) -> char* {
    char* p = ws + off;
    off = (off + bytes + 255) & ~(size_t)255;
    return p;
  };
  const long NP = 65536;  // pixels in tp1 order
  float* t_h = (float*)take(NP * 16 * 4);
  float* t_w = (float*)take(NP * 16 * 4);
  float* z_h = (float*)take(NP * 16 * 4);
  float* z_w = (float*)take(NP * 16 * 4);
  unsigned short* h_bf = (unsigned short*)take(NP * 512 * 2);
  unsigned short* w_bf = (unsigned short*)take(NP * 512 * 2);
  unsigned short* c_bf = (unsigned short*)take(NP * 512 * 2);
  unsigned short* y_bf = (unsigned short*)take(NP * 512 * 2);
  unsigned short* pinT  = (unsigned short*)take(32 * 512 * 2);
  unsigned short* fullH = (unsigned short*)take(128 * 1024 * 2);
  unsigned short* fullW = (unsigned short*)take(128 * 1024 * 2);
  unsigned short* poutT = (unsigned short*)take(2 * 512 * 16 * 2);
  unsigned short* pwT   = (unsigned short*)take(512 * 512 * 2);
  float* a_sum = (float*)take(16 * 512 * 4);
  float* wgt   = (float*)take(3 * 16 * 512 * 4);

  (void)in_sizes; (void)n_in; (void)out_size; (void)ws_size;

  k_prep<<<512, 256, 0, stream>>>(mh_pin_w, mw_pin_w, mh_full_w, mw_full_w,
                                  mh_pout_w, mw_pout_w, proj_w,
                                  pinT, fullH, fullW, poutT, pwT, a_sum);

  k_proj_in<<<4096, 64, 0, stream>>>(x, pinT, mh_pin_b, mw_pin_b, t_h, t_w);

  k_mix<<<dim3(64, 16), 128, 0, stream>>>(t_h, fullH, mh_full_b, z_h, 0);
  k_mix<<<dim3(64, 16), 128, 0, stream>>>(t_w, fullW, mw_full_b, z_w, 1);

  k_proj_out<<<4096, 64, 0, stream>>>(z_h, z_w, poutT, mh_pout_b, mw_pout_b,
                                      h_bf, w_bf);

  k_mixc<<<(16 * 32 * 32 * 512) / 256, 256, 0, stream>>>(x, c_weight, c_bias,
                                                         c_bf);

  k_mean<<<256, 512, 0, stream>>>(h_bf, w_bf, c_bf, a_sum);

  k_reweight<<<16, 128, 0, stream>>>(a_sum, rw_fc1_w, rw_fc1_b,
                                     rw_fc2_w, rw_fc2_b, wgt);

  k_combine<<<131072, 256, 0, stream>>>(h_bf, w_bf, c_bf, wgt, y_bf);

  k_final<<<dim3(4096, 2), 128, 0, stream>>>(y_bf, pwT, proj_b, out);
}
